// MusicLSTM_43130061586815
// MI455X (gfx1250) — compile-verified, matrix operand order checked by reading, not run on hardware
//
#include <hip/hip_runtime.h>
#include <hip/hip_bf16.h>

// MusicLSTM fused kernel for gfx1250 (MI455X).
// B=256, T=2048, I=2, H=128, P=129.
// Grid: 16 workgroups (one 16-row batch tile each), 512 threads (16 waves).
// waves 0..15: 2 gate N-tiles each of the recurrent GEMM h @ W_hh^T (4H=512 cols)
// waves 0..8 : 1 head N-tile each (note cols 0..143 padded; col 128 = note[128],
//              col 129 = duration head). Heads reuse the A fragments of the
//              recurrent GEMM and write masked+biased results directly to out.

typedef __attribute__((ext_vector_type(16))) _Float16 v16h;
typedef __attribute__((ext_vector_type(8)))  _Float16 v8h;
typedef __attribute__((ext_vector_type(8)))  float    v8f;

#define B_ 256
#define T_ 2048
#define H_ 128
#define P_ 129

__device__ __forceinline__ float fsigm(float v) {
    return __builtin_amdgcn_rcpf(1.0f + __expf(-v));
}
__device__ __forceinline__ float ftanh(float v) {
    // tanh(v) = 1 - 2/(1+e^{2v})
    return 1.0f - 2.0f * __builtin_amdgcn_rcpf(1.0f + __expf(2.0f * v));
}

__device__ __forceinline__ v8h cvt8(float4 a, float4 b) {
    v8h r;
    r[0] = (_Float16)a.x; r[1] = (_Float16)a.y; r[2] = (_Float16)a.z; r[3] = (_Float16)a.w;
    r[4] = (_Float16)b.x; r[5] = (_Float16)b.y; r[6] = (_Float16)b.z; r[7] = (_Float16)b.w;
    return r;
}

#define JOIN16(lo, hi) __builtin_shufflevector(lo, hi, 0,1,2,3,4,5,6,7,8,9,10,11,12,13,14,15)

// Opaque register pin: makes the f16 fragment asm-defined so the register
// allocator cannot rematerialize the f32->f16 conversion inside the time loop
// (which would keep the f32 weights live and burn ~96 v_cvt per iteration).
#define PIN(v) asm volatile("" : "+v"(v))

#define WMMA_F16(A, Bm, C) \
    __builtin_amdgcn_wmma_f32_16x16x32_f16(false, (A), false, (Bm), (short)0, (C), false, false)

__global__ __launch_bounds__(512, 1)
void music_lstm_fused(const float* __restrict__ x,        // [B,T,2]
                      const int*   __restrict__ lengths,  // [B]
                      const float* __restrict__ W_ih,     // [512,2]
                      const float* __restrict__ W_hh,     // [512,128]
                      const float* __restrict__ note_W,   // [129,128]
                      const float* __restrict__ note_b,   // [129]
                      const float* __restrict__ dur_W,    // [1,128]
                      const float* __restrict__ dur_b,    // [1]
                      float* __restrict__ out)            // note|dur|hT|cT concat
{
    // LDS state for the batch tile
    __shared__ _Float16 hA[16][136];   // h state, f16, WMMA A-fragment friendly
    __shared__ float    Gs[16][516];   // gate pre-activations (h @ W_hh^T)
    __shared__ float    cS[16][128];   // cell state (f32)
    __shared__ float    hF[16][128];   // h state (f32, for hT output)
    __shared__ float    Wih_s[512][2];
    __shared__ int      len_s[16];

    const int tid  = threadIdx.x;
    const int lane = tid & 31;
    const int w    = tid >> 5;          // wave id 0..15
    const int m16  = lane & 15;
    const int hi8  = (lane >> 4) << 3;  // 0 for lanes 0-15, 8 for lanes 16-31
    const int b0   = blockIdx.x * 16;   // batch tile base

    // ---- init LDS state ----
    for (int i = tid; i < 16 * 128; i += 512) {
        int m = i >> 7, j = i & 127;
        cS[m][j] = 0.0f;
        hF[m][j] = 0.0f;
        hA[m][j] = (_Float16)0.0f;
    }
    for (int i = tid; i < 1024; i += 512) Wih_s[i >> 1][i & 1] = W_ih[i];
    if (tid < 16) len_s[tid] = lengths[b0 + tid];

    // ---- preload recurrent B fragments (W_hh, f32 -> f16) into VGPRs ----
    // B fragment layout mirrors documented 16-bit A layout over the [N][K]
    // row-major weight matrix (gates[m][n] = sum_k h[m][k] * W_hh[n][k]).
    v16h bg[2][4];
#pragma unroll
    for (int i = 0; i < 2; ++i) {
        const int n = 32 * w + 16 * i + m16;
        const float* wr = W_hh + (size_t)n * H_;
#pragma unroll
        for (int kc = 0; kc < 4; ++kc) {
            const int kb = kc * 32 + hi8;
            v8h lo = cvt8(*(const float4*)(wr + kb),      *(const float4*)(wr + kb + 4));
            v8h hi = cvt8(*(const float4*)(wr + kb + 16), *(const float4*)(wr + kb + 20));
            bg[i][kc] = JOIN16(lo, hi);
            PIN(bg[i][kc]);
        }
    }

    // ---- preload head B fragments (note_W rows 0..128, dur_W as col 129) ----
    v16h bh[4];
    float hbias = 0.0f;
    const int hn = 16 * w + m16;   // head output column for waves 0..8
    if (w < 9) {
        const float* src = (hn < P_) ? (note_W + (size_t)hn * H_)
                                     : ((hn == P_) ? dur_W : (const float*)0);
#pragma unroll
        for (int kc = 0; kc < 4; ++kc) {
            v8h lo = {}, hi = {};
            if (src) {
                const int kb = kc * 32 + hi8;
                lo = cvt8(*(const float4*)(src + kb),      *(const float4*)(src + kb + 4));
                hi = cvt8(*(const float4*)(src + kb + 16), *(const float4*)(src + kb + 20));
            }
            bh[kc] = JOIN16(lo, hi);
        }
        if (hn < P_)       hbias = note_b[hn];
        else if (hn == P_) hbias = dur_b[0];
    } else {
#pragma unroll
        for (int kc = 0; kc < 4; ++kc) bh[kc] = v16h{};
    }
#pragma unroll
    for (int kc = 0; kc < 4; ++kc) PIN(bh[kc]);

    __syncthreads();

    float* outN = out;                               // [B,T,129]
    float* outD = out + (size_t)B_ * T_ * P_;        // [B,T]
    float* outH = outD + (size_t)B_ * T_;            // [B,128]
    float* outC = outH + (size_t)B_ * H_;            // [B,128]

    const _Float16* hrow = &hA[m16][0];

    for (int k = 0; k < T_; ++k) {
        // ================= phase A: WMMA =================
        // Load A fragments of h_{k-1} (two ds_load_b128 per K-chunk).
        v16h a[4];
#pragma unroll
        for (int kc = 0; kc < 4; ++kc) {
            const int kb = kc * 32 + hi8;
            v8h lo = *(const v8h*)(hrow + kb);
            v8h hi = *(const v8h*)(hrow + kb + 16);
            a[kc] = JOIN16(lo, hi);
        }

        // Recurrent gates: 2 N-tiles per wave, K chain of 4.
        v8f cg[2];
        cg[0] = v8f{};
        cg[1] = v8f{};
#pragma unroll
        for (int kc = 0; kc < 4; ++kc) {
            cg[0] = WMMA_F16(a[kc], bg[0][kc], cg[0]);
            cg[1] = WMMA_F16(a[kc], bg[1][kc], cg[1]);
        }

        // Output heads for timestep k-1 (h_{k-1} is what the reference emits).
        if (w < 9) {
            v8f ch = v8f{};
#pragma unroll
            for (int kc = 0; kc < 4; ++kc) ch = WMMA_F16(a[kc], bh[kc], ch);
            if (k > 0) {
                const int t = k - 1;
#pragma unroll
                for (int r = 0; r < 8; ++r) {
                    const int m  = r + hi8;
                    const int gb = b0 + m;
                    const bool valid = t < len_s[m];
                    const float v = valid ? (ch[r] + hbias) : 0.0f;
                    if (hn < P_)
                        __builtin_nontemporal_store(v, &outN[((size_t)gb * T_ + t) * P_ + hn]);
                    else if (hn == P_)
                        __builtin_nontemporal_store(v, &outD[(size_t)gb * T_ + t]);
                }
            }
        }

        // Scatter gate pre-activations to LDS per documented C layout.
#pragma unroll
        for (int i = 0; i < 2; ++i) {
            const int gcol = 32 * w + 16 * i + m16;
#pragma unroll
            for (int r = 0; r < 8; ++r) Gs[r + hi8][gcol] = cg[i][r];
        }

        __syncthreads();

        // ================= phase B: elementwise LSTM update =================
#pragma unroll
        for (int e = 0; e < 4; ++e) {
            const int idx = tid + 512 * e;      // 0..2047
            const int m = idx >> 7;
            const int j = idx & 127;

            const float2 xv = *(const float2*)(x + ((size_t)(b0 + m) * T_ + k) * 2);
            const float x0 = xv.x, x1 = xv.y;

            const float2 wi = *(const float2*)&Wih_s[j][0];
            const float2 wf = *(const float2*)&Wih_s[j + 128][0];
            const float2 wg = *(const float2*)&Wih_s[j + 256][0];
            const float2 wo = *(const float2*)&Wih_s[j + 384][0];

            const float gi = Gs[m][j]       + wi.x * x0 + wi.y * x1;
            const float gf = Gs[m][j + 128] + wf.x * x0 + wf.y * x1;
            const float gg = Gs[m][j + 256] + wg.x * x0 + wg.y * x1;
            const float go = Gs[m][j + 384] + wo.x * x0 + wo.y * x1;

            const float ig = fsigm(gi);
            const float fg = fsigm(gf);
            const float og = fsigm(go);
            const float g  = ftanh(gg);

            const float cold = cS[m][j];
            const float cn   = fg * cold + ig * g;
            const float hnw  = og * ftanh(cn);

            const bool valid = k < len_s[m];
            const float hsel = valid ? hnw : hF[m][j];
            const float csel = valid ? cn  : cold;

            cS[m][j] = csel;
            hF[m][j] = hsel;
            hA[m][j] = (_Float16)hsel;
        }

        __syncthreads();
    }

    // ---- epilogue: heads for the final timestep t = T-1 ----
    {
        v16h a[4];
#pragma unroll
        for (int kc = 0; kc < 4; ++kc) {
            const int kb = kc * 32 + hi8;
            v8h lo = *(const v8h*)(hrow + kb);
            v8h hi = *(const v8h*)(hrow + kb + 16);
            a[kc] = JOIN16(lo, hi);
        }
        if (w < 9) {
            v8f ch = v8f{};
#pragma unroll
            for (int kc = 0; kc < 4; ++kc) ch = WMMA_F16(a[kc], bh[kc], ch);
            const int t = T_ - 1;
#pragma unroll
            for (int r = 0; r < 8; ++r) {
                const int m  = r + hi8;
                const int gb = b0 + m;
                const bool valid = t < len_s[m];
                const float v = valid ? (ch[r] + hbias) : 0.0f;
                if (hn < P_)
                    __builtin_nontemporal_store(v, &outN[((size_t)gb * T_ + t) * P_ + hn]);
                else if (hn == P_)
                    __builtin_nontemporal_store(v, &outD[(size_t)gb * T_ + t]);
            }
        }
    }

    // ---- final states hT, cT (frozen at last valid step) ----
#pragma unroll
    for (int e = 0; e < 4; ++e) {
        const int idx = tid + 512 * e;
        const int m = idx >> 7;
        const int j = idx & 127;
        __builtin_nontemporal_store(hF[m][j], &outH[(size_t)(b0 + m) * H_ + j]);
        __builtin_nontemporal_store(cS[m][j], &outC[(size_t)(b0 + m) * H_ + j]);
    }
}

extern "C" void kernel_launch(void* const* d_in, const int* in_sizes, int n_in,
                              void* d_out, int out_size, void* d_ws, size_t ws_size,
                              hipStream_t stream) {
    (void)in_sizes; (void)n_in; (void)out_size; (void)d_ws; (void)ws_size;
    const float* x       = (const float*)d_in[0];
    const int*   lengths = (const int*)  d_in[1];
    const float* W_ih    = (const float*)d_in[2];
    const float* W_hh    = (const float*)d_in[3];
    const float* note_W  = (const float*)d_in[4];
    const float* note_b  = (const float*)d_in[5];
    const float* dur_W   = (const float*)d_in[6];
    const float* dur_b   = (const float*)d_in[7];
    float* out = (float*)d_out;

    music_lstm_fused<<<dim3(B_ / 16), dim3(512), 0, stream>>>(
        x, lengths, W_ih, W_hh, note_W, note_b, dur_W, dur_b, out);
}
